// MLSTMCell_67070209295132
// MI455X (gfx1250) — matrix operand order repaired and checked by reading, not compiled, not run on hardware
//
#include <hip/hip_runtime.h>
#include <stdint.h>

// ---------------- problem sizes (compile-time) ----------------
constexpr int B_DIM = 4096;
constexpr int I_DIM = 512;
constexpr int H_DIM = 512;
constexpr int O_DIM = 512;
constexpr int K_CELL = 32;

constexpr int XK = I_DIM + 2 * H_DIM;   // 1536 packed input width (sample|hidden|mem_para)
constexpr int GN = 4 * H_DIM;           // 2048 packed gate output width (i,o,c,m)

// ---------------- vector types ----------------
typedef __attribute__((ext_vector_type(16))) __bf16         v16bf;
typedef __attribute__((ext_vector_type(8)))  float          v8f;
typedef __attribute__((ext_vector_type(8)))  unsigned short us8;
typedef __attribute__((ext_vector_type(16))) unsigned short us16;

// float -> bf16 with round-to-nearest-even
__device__ __forceinline__ unsigned short f2bf(float f) {
    unsigned int u = __float_as_uint(f);
    u += 0x7fffu + ((u >> 16) & 1u);
    return (unsigned short)(u >> 16);
}

__device__ __forceinline__ float sigmoidf_(float x) {
    return 1.0f / (1.0f + expf(-x));
}

// gfx1250 async global->LDS copy (16B per lane), tracked by ASYNCcnt.
__device__ __forceinline__ void async_copy_b128(unsigned lds_off, const void* gaddr) {
    asm volatile("global_load_async_to_lds_b128 %0, %1, off"
                 :: "v"(lds_off), "v"((unsigned long long)(uintptr_t)gaddr)
                 : "memory");
}
__device__ __forceinline__ void wait_async_le3() {
    asm volatile("s_wait_asynccnt 0x3" ::: "memory");
}
__device__ __forceinline__ void wait_async_0() {
    asm volatile("s_wait_asynccnt 0x0" ::: "memory");
}

// ---------------- pack kernels ----------------
__global__ __launch_bounds__(256)
void pack_x_kernel(const float* __restrict__ s, const float* __restrict__ h,
                   const float* __restrict__ mp, unsigned short* __restrict__ X) {
    size_t i = (size_t)blockIdx.x * 256 + threadIdx.x;    // < B_DIM * XK
    int c = (int)(i % XK);
    size_t b = i / XK;
    float v;
    if (c < I_DIM)              v = s[b * I_DIM + c];
    else if (c < I_DIM + H_DIM) v = h[b * H_DIM + (c - I_DIM)];
    else                        v = mp[b * H_DIM + (c - I_DIM - H_DIM)];
    X[i] = f2bf(v);
}

__global__ __launch_bounds__(256)
void pack_w_kernel(const float* __restrict__ Wi, const float* __restrict__ Wo,
                   const float* __restrict__ Wc, const float* __restrict__ Wm,
                   unsigned short* __restrict__ W) {
    size_t i = (size_t)blockIdx.x * 256 + threadIdx.x;    // < GN * XK
    int c   = (int)(i % XK);
    int row = (int)(i / XK);                              // 0..2047
    float v;
    if (row < 3 * H_DIM) {                                // i, o, c gates (K = I+H = 1024)
        int g = row >> 9;                                 // 0,1,2
        int hh = row & (H_DIM - 1);
        const float* Wg = (g == 0) ? Wi : (g == 1) ? Wo : Wc;
        v = (c < I_DIM + H_DIM) ? Wg[(size_t)hh * (I_DIM + H_DIM) + c] : 0.0f;
    } else {                                              // m gate (K = I+2H = 1536)
        int hh = row - 3 * H_DIM;
        v = Wm[(size_t)hh * XK + c];
    }
    W[i] = f2bf(v);
}

__global__ __launch_bounds__(256)
void pack_bias_kernel(const float* __restrict__ bi, const float* __restrict__ bo,
                      const float* __restrict__ bc, const float* __restrict__ bm,
                      float* __restrict__ bcat) {
    int i = blockIdx.x * 256 + threadIdx.x;               // < GN
    int g = i >> 9, hh = i & (H_DIM - 1);
    const float* bg = (g == 0) ? bi : (g == 1) ? bo : (g == 2) ? bc : bm;
    bcat[i] = bg[hh];
}

__global__ __launch_bounds__(256)
void pack_wout_kernel(const float* __restrict__ W, unsigned short* __restrict__ Wb) {
    size_t i = (size_t)blockIdx.x * 256 + threadIdx.x;    // < O_DIM * H_DIM
    Wb[i] = f2bf(W[i]);
}

// ---------------- WMMA bf16 GEMM (C = A * B^T + bias), NT layout ----------------
// A: [M, Kd] bf16 row-major, Bm: [N, Kd] bf16 row-major, C: [M, N] f32.
// Block: 256 threads = 8 waves -> 128(M) x 64(N) tile, wave = 32x32 (2x2 WMMA).
// Double-buffered LDS fed by GLOBAL_LOAD_ASYNC_TO_LDS_B128 (ASYNCcnt pipeline).
constexpr int TSTRIDE = 40;                   // padded LDS row stride (halfs)
constexpr int ABUF_HALFS = 128 * TSTRIDE;     // per-buffer A tile
constexpr int BBUF_HALFS = 64 * TSTRIDE;      // per-buffer B tile

__global__ __launch_bounds__(256)
void gemm_bf16_nt(const unsigned short* __restrict__ A,
                  const unsigned short* __restrict__ Bm,
                  const float* __restrict__ bias,
                  float* __restrict__ C,
                  int M, int N, int Kd) {
    __shared__ unsigned short At[2][ABUF_HALFS];
    __shared__ unsigned short Bt[2][BBUF_HALFS];

    const int tid  = threadIdx.x;
    const int lane = tid & 31;
    const int wave = tid >> 5;
    const int wm = wave >> 1;                 // 0..3 -> M sub-tile
    const int wn = wave & 1;                  // 0..1 -> N sub-tile
    const int tileM = blockIdx.y * 128;
    const int tileN = blockIdx.x * 64;
    const int hi = lane >> 4;
    const int l16 = lane & 15;

    // --- per-thread async copy slots (3 x 16B per tile) ---
    const int rowA0 = tid >> 2;               // chunk c = tid        -> rows 0..63
    const int rowA1 = rowA0 + 64;             // chunk c = tid + 256  -> rows 64..127
    const int colC  = (tid & 3) << 3;         // 0,8,16,24 (halfs)
    const unsigned ldsA0 = (unsigned)(uintptr_t)&At[0][rowA0 * TSTRIDE + colC];
    const unsigned ldsA1 = (unsigned)(uintptr_t)&At[0][rowA1 * TSTRIDE + colC];
    const unsigned ldsB0 = (unsigned)(uintptr_t)&Bt[0][rowA0 * TSTRIDE + colC];
    const unsigned short* gA0 = A  + (size_t)(tileM + rowA0) * Kd + colC;
    const unsigned short* gA1 = A  + (size_t)(tileM + rowA1) * Kd + colC;
    const unsigned short* gB0 = Bm + (size_t)(tileN + rowA0) * Kd + colC;

    auto issue_tile = [&](int kt, int buf) {
        const unsigned aoff = (unsigned)(buf * ABUF_HALFS * 2);
        const unsigned boff = (unsigned)(buf * BBUF_HALFS * 2);
        const int kh = kt << 5;               // K offset in halfs
        async_copy_b128(ldsA0 + aoff, gA0 + kh);
        async_copy_b128(ldsA1 + aoff, gA1 + kh);
        async_copy_b128(ldsB0 + boff, gB0 + kh);
    };

    v8f acc[2][2] = {};

    const int ksteps = Kd >> 5;
    issue_tile(0, 0);                          // prologue prefetch

    for (int kt = 0; kt < ksteps; ++kt) {
        const int buf = kt & 1;
        if (kt + 1 < ksteps) {
            issue_tile(kt + 1, buf ^ 1);       // overlap next tile's DMA with compute
            wait_async_le3();                  // current tile's 3 copies complete (in-order)
        } else {
            wait_async_0();
        }
        __syncthreads();                       // copies visible to all 8 waves

        // A fragment (16x32 bf16): lane l16 = row M, hi selects K halves per ISA layout:
        //   k(j) = (j%8) + 8*hi + 16*(j/8) -> two contiguous 8-half chunks at +8*hi, +16+8*hi
        v16bf afrag[2];
        #pragma unroll
        for (int fr = 0; fr < 2; ++fr) {
            int m = wm * 32 + fr * 16 + l16;
            const us8* p = reinterpret_cast<const us8*>(&At[buf][m * TSTRIDE + 8 * hi]);
            us8 c0 = p[0];
            us8 c1 = p[2];   // +16 halfs
            us16 t = __builtin_shufflevector(c0, c1, 0,1,2,3,4,5,6,7,8,9,10,11,12,13,14,15);
            afrag[fr] = __builtin_bit_cast(v16bf, t);
        }
        // B fragment (32x16 bf16): lane l16 = col N, k(j) = (j%8) + 8*hi + 16*(j/8) (same as A)
        v16bf bfrag[2];
        #pragma unroll
        for (int fc = 0; fc < 2; ++fc) {
            int n = wn * 32 + fc * 16 + l16;
            const us8* p = reinterpret_cast<const us8*>(&Bt[buf][n * TSTRIDE + 8 * hi]);
            us8 d0 = p[0];
            us8 d1 = p[2];
            us16 t = __builtin_shufflevector(d0, d1, 0,1,2,3,4,5,6,7,8,9,10,11,12,13,14,15);
            bfrag[fc] = __builtin_bit_cast(v16bf, t);
        }
        #pragma unroll
        for (int fr = 0; fr < 2; ++fr)
            #pragma unroll
            for (int fc = 0; fc < 2; ++fc)
                acc[fr][fc] = __builtin_amdgcn_wmma_f32_16x16x32_bf16(
                    false, afrag[fr], false, bfrag[fc],
                    (short)0, acc[fr][fc], false, false);

        __syncthreads();                       // readers done before buf is overwritten
    }

    // Epilogue: C/D layout -> VGPR r, lanes 0-15: M=r, lanes 16-31: M=r+8; N=lane%16
    #pragma unroll
    for (int fr = 0; fr < 2; ++fr) {
        #pragma unroll
        for (int fc = 0; fc < 2; ++fc) {
            int n = tileN + wn * 32 + fc * 16 + l16;
            float bv = bias[n];
            int mBase = tileM + wm * 32 + fr * 16 + hi * 8;
            #pragma unroll
            for (int r = 0; r < 8; ++r)
                C[(size_t)(mBase + r) * N + n] = acc[fr][fc][r] + bv;
        }
    }
}

// ---------------- fused gate / memory-weight / cell-shift kernel ----------------
// One read + one write per cell_tensors element (HBM roofline floor).
__global__ __launch_bounds__(256)
void cell_update_kernel(const float* __restrict__ preact,
                        const float* __restrict__ cells,
                        float* __restrict__ out,
                        unsigned short* __restrict__ hiddenb) {
    const size_t BH = (size_t)B_DIM * H_DIM;
    size_t idx = (size_t)blockIdx.x * 256 + threadIdx.x;  // < B*H
    size_t b = idx >> 9;
    int h = (int)(idx & (H_DIM - 1));

    const float* pre = preact + b * GN;
    float gi = sigmoidf_(pre[0 * H_DIM + h]);
    float go = sigmoidf_(pre[1 * H_DIM + h]);
    float pc = tanhf(pre[2 * H_DIM + h]);
    float gm = 0.5f * sigmoidf_(pre[3 * H_DIM + h]);

    float* out_hidden = out + (size_t)B_DIM * O_DIM;
    float* out_cells  = out_hidden + BH;
    float* out_mem    = out_cells + (size_t)K_CELL * BH;

    // cell_filtered = -sum_j cells[jj]*p[K-1-jj]; fused with the shift copy.
    float p = 1.0f, acc = 0.0f;
    #pragma unroll
    for (int j = 0; j < K_CELL; ++j) {
        p *= ((float)j - gm) * (1.0f / (float)(j + 1));   // cumprod factor
        int jj = K_CELL - 1 - j;
        float cv = cells[(size_t)jj * BH + idx];
        acc = fmaf(cv, p, acc);
        if (jj >= 1)
            out_cells[(size_t)(jj - 1) * BH + idx] = cv;  // updated[0..K-2] = cells[1..K-1]
    }
    float cnew = fmaf(pc, gi, -acc);
    out_cells[(size_t)(K_CELL - 1) * BH + idx] = cnew;

    float hn = go * tanhf(cnew);
    out_hidden[idx] = hn;
    out_mem[idx] = gm;
    hiddenb[idx] = f2bf(hn);
}

// ---------------- host launch ----------------
extern "C" void kernel_launch(void* const* d_in, const int* in_sizes, int n_in,
                              void* d_out, int out_size, void* d_ws, size_t ws_size,
                              hipStream_t stream) {
    const float* sample = (const float*)d_in[0];
    const float* hidden = (const float*)d_in[1];
    const float* cells  = (const float*)d_in[2];
    const float* memp   = (const float*)d_in[3];
    const float* W_i   = (const float*)d_in[4];
    const float* b_i   = (const float*)d_in[5];
    const float* W_o   = (const float*)d_in[6];
    const float* b_o   = (const float*)d_in[7];
    const float* W_m   = (const float*)d_in[8];
    const float* b_m   = (const float*)d_in[9];
    const float* W_c   = (const float*)d_in[10];
    const float* b_c   = (const float*)d_in[11];
    const float* W_out = (const float*)d_in[12];
    const float* b_out = (const float*)d_in[13];
    float* out = (float*)d_out;

    char* ws = (char*)d_ws;
    size_t off = 0;
    unsigned short* Xcat = (unsigned short*)(ws + off); off += (size_t)B_DIM * XK * 2;      // 12.6 MB
    unsigned short* Wcat = (unsigned short*)(ws + off); off += (size_t)GN * XK * 2;         //  6.3 MB
    unsigned short* Woutb = (unsigned short*)(ws + off); off += (size_t)O_DIM * H_DIM * 2;  //  0.5 MB
    unsigned short* hiddenb = (unsigned short*)(ws + off); off += (size_t)B_DIM * H_DIM * 2;//  4.2 MB
    float* bcat = (float*)(ws + off); off += (size_t)GN * 4;                                //  8 KB
    float* preact = (float*)(ws + off); off += (size_t)B_DIM * GN * 4;                      // 33.6 MB

    pack_x_kernel<<<(B_DIM * XK) / 256, 256, 0, stream>>>(sample, hidden, memp, Xcat);
    pack_w_kernel<<<(GN * XK) / 256, 256, 0, stream>>>(W_i, W_o, W_c, W_m, Wcat);
    pack_bias_kernel<<<GN / 256, 256, 0, stream>>>(b_i, b_o, b_c, b_m, bcat);
    pack_wout_kernel<<<(O_DIM * H_DIM) / 256, 256, 0, stream>>>(W_out, Woutb);

    dim3 g1(GN / 64, B_DIM / 128);
    gemm_bf16_nt<<<g1, 256, 0, stream>>>(Xcat, Wcat, bcat, preact, B_DIM, GN, XK);

    cell_update_kernel<<<(B_DIM * H_DIM) / 256, 256, 0, stream>>>(preact, cells, out, hiddenb);

    dim3 g2(O_DIM / 64, B_DIM / 128);
    gemm_bf16_nt<<<g2, 256, 0, stream>>>(hiddenb, Woutb, b_out, out, B_DIM, O_DIM, H_DIM);
}
